// HierarchicalExplainableBearingNet_28114855920244
// MI455X (gfx1250) — matrix operand order repaired and checked by reading, not compiled, not run on hardware
//
#include <hip/hip_runtime.h>
#include <hip/hip_bf16.h>
#include <math.h>

typedef __attribute__((ext_vector_type(16))) __bf16 v16bf;
typedef __attribute__((ext_vector_type(8)))  float  v8f;

constexpr int B_ = 256, S_ = 3, L_ = 4096, C_ = 2, G_ = 768;

// ============================ prep kernels ============================

// src: x viewed as [G][L][C] f32 -> dst [G][C][L] bf16
__global__ __launch_bounds__(256) void cast_transpose_kernel(const float* __restrict__ src,
                                                             __bf16* __restrict__ dst) {
  size_t idx = (size_t)blockIdx.x * blockDim.x + threadIdx.x;
  size_t total = (size_t)G_ * C_ * L_;
  if (idx >= total) return;
  int g = (int)(idx / (C_ * L_));
  int r = (int)(idx % (C_ * L_));
  int c = r / L_;
  int l = r % L_;
  dst[idx] = (__bf16)src[((size_t)g * L_ + l) * C_ + c];
}

// Swizzle conv weights into WMMA A-fragment order:
// dst[br][ct][q][lane][j] = w[br][co = ct*16 + (lane&15)]
//                            [k = q*32 + 8*(lane>>4) + (j<8 ? j : j+8)]  (0 if k >= cink)
// so each lane's 16 bf16 A-fragment values are contiguous (32B) in memory.
__global__ __launch_bounds__(256) void pack_w_swizzle_kernel(const float* __restrict__ src,
                                                             __bf16* __restrict__ dst,
                                                             int cout, int cink, int kp) {
  int idx = blockIdx.x * blockDim.x + threadIdx.x;
  int per_br = cout * kp;
  if (idx >= 2 * per_br) return;
  int br = idx / per_br, e = idx - br * per_br;
  int j = e & 15;
  int lane = (e >> 4) & 31;
  int rest = e >> 9;             // = ct*(kp/32) + q
  int kpq = kp >> 5;
  int q = rest % kpq;
  int ct = rest / kpq;
  int co = ct * 16 + (lane & 15);
  int k = q * 32 + ((lane >> 4) << 3) + (j < 8 ? j : j + 8);
  __bf16 v = (__bf16)0.0f;
  if (k < cink) v = (__bf16)src[((size_t)br * cout + co) * cink + k];
  dst[idx] = v;
}

// fold conv bias + eval-mode BN into scale/shift
__global__ void bn_prep_kernel(const float* cb, const float* g, const float* be,
                               const float* m, const float* v,
                               float* scale, float* shift, int n) {
  int i = blockIdx.x * blockDim.x + threadIdx.x;
  if (i >= n) return;
  float s = g[i] * rsqrtf(v[i] + 1e-5f);
  scale[i] = s;
  shift[i] = be[i] + (cb[i] - m[i]) * s;
}

__global__ void zero_kernel(float* p, int n) {
  int i = blockIdx.x * blockDim.x + threadIdx.x;
  if (i < n) p[i] = 0.0f;
}

// ====================== WMMA implicit-GEMM conv =======================
// Block: (g, 64-wide L tile). One wave per 16-row Cout tile, NT=4 accumulators.
// Phase 1: raw input window -> LDS. Phase 2: expand to B-fragment-order LDS.
// Main loop: A = 1 contiguous 32B global load (pre-swizzled), reused over NT
// WMMAs; each B fragment = 1 contiguous 32B LDS load.
template <int CIN, int COUT, int KW, int STRIDE, int PAD, int LIN, int LOUT, int KP,
          int NT, bool POOL>
__global__ __launch_bounds__(32 * (COUT / 16)) void conv_wmma_kernel(
    const __bf16* __restrict__ xin,   // [G][CIN][LIN]
    const __bf16* __restrict__ wA,    // A-fragment swizzled [COUT/16][KP/32][32][16]
    const float* __restrict__ scale,  // [COUT]
    const float* __restrict__ shift,  // [COUT]
    __bf16* __restrict__ out,         // [G][COUT][LOUT]  (if !POOL)
    float* __restrict__ pool_out,     // base + g*pool_stride + co (if POOL)
    int pool_stride, float pool_scale) {
  constexpr int NW = COUT / 16;
  constexpr int KPQ = KP / 32;
  constexpr int LT = 16 * NT;                   // output columns per block
  constexpr int WLEN = (LT - 1) * STRIDE + KW;  // raw input window length
  __shared__ __align__(16) __bf16 sIn[CIN * WLEN];
  __shared__ __align__(32) __bf16 sB[KPQ * NT * 512];  // [q][t][lane][j]

  const int g = blockIdx.x;
  const int l0 = blockIdx.y * LT;
  const int tid = threadIdx.x;
  constexpr int NTHR = 32 * NW;
  const __bf16* xg = xin + (size_t)g * CIN * LIN;

  // phase 1: cooperative raw-window load (zero padded at edges)
  for (int i = tid; i < CIN * WLEN; i += NTHR) {
    int ci = i / WLEN, off = i - ci * WLEN;
    int pos = l0 * STRIDE - PAD + off;
    __bf16 v = (__bf16)0.0f;
    if (pos >= 0 && pos < LIN) v = xg[(size_t)ci * LIN + pos];
    sIn[i] = v;
  }
  __syncthreads();

  // phase 2: im2col expand into B-fragment order
  // element (kk = q*32 + 8*(lane>>4) + (j<8 ? j : j+8), n = t*16 + (lane&15))
  for (int r = tid; r < KPQ * NT * 512; r += NTHR) {
    int j = r & 15;
    int ln = (r >> 4) & 31;
    int qt = r >> 9;            // q*NT + t
    int t = qt % NT;
    int q = qt / NT;
    int kk = q * 32 + ((ln >> 4) << 3) + (j < 8 ? j : j + 8);
    __bf16 v = (__bf16)0.0f;
    if (kk < CIN * KW) {
      int ci = kk / KW, kw = kk - ci * KW;
      int n = t * 16 + (ln & 15);
      v = sIn[ci * WLEN + n * STRIDE + kw];
    }
    sB[r] = v;
  }
  __syncthreads();

  const int lane = tid & 31;
  const int half = lane >> 4;
  const int mn = lane & 15;
  const int ct = tid >> 5;  // Cout tile of this wave

  v8f zero = {};
  v8f acc[NT];
#pragma unroll
  for (int t = 0; t < NT; ++t) acc[t] = zero;

  const v16bf* ap = (const v16bf*)wA;
  for (int q = 0; q < KPQ; ++q) {
    v16bf a = ap[(ct * KPQ + q) * 32 + lane];  // 32B contiguous, reused NT times
#pragma unroll
    for (int t = 0; t < NT; ++t) {
      v16bf b = *(const v16bf*)(sB + (((q * NT + t) * 32 + lane) << 4));
      acc[t] = __builtin_amdgcn_wmma_f32_16x16x32_bf16(false, a, false, b,
                                                       (short)0, acc[t], false, false);
    }
  }

  // epilogue: acc[t][i] holds (M = i + 8*half, N = t*16 + mn)
#pragma unroll
  for (int i = 0; i < 8; ++i) {
    int co = ct * 16 + i + 8 * half;
    float sc_ = scale[co], sh_ = shift[co];
    if constexpr (POOL) {
      float ysum = 0.0f;
#pragma unroll
      for (int t = 0; t < NT; ++t) ysum += fmaxf(fmaf(acc[t][i], sc_, sh_), 0.0f);
#pragma unroll
      for (int off = 1; off < 16; off <<= 1) ysum += __shfl_xor(ysum, off, 32);
      if (mn == 0) atomicAdd(pool_out + (size_t)g * pool_stride + co, ysum * pool_scale);
    } else {
#pragma unroll
      for (int t = 0; t < NT; ++t) {
        float y = fmaxf(fmaf(acc[t][i], sc_, sh_), 0.0f);
        out[((size_t)g * COUT + co) * LOUT + l0 + t * 16 + mn] = (__bf16)y;
      }
    }
  }
}

// ============================ GAT (4-node) ============================
__global__ __launch_bounds__(256) void gat_kernel(
    const float* __restrict__ feat,     // [G][256]
    const float* __restrict__ lin_w,    // [256][64]
    const float* __restrict__ att_src,  // [4][64]
    const float* __restrict__ att_dst,  // [4][64]
    const float* __restrict__ bias,     // [256]
    float* __restrict__ gout) {         // [G][1024]
  __shared__ float fs[256];
  __shared__ float hs[1024];
  __shared__ float as[16], ad[16];
  __shared__ float alpha[16][4];
  int g = blockIdx.x, t = threadIdx.x;
  fs[t] = feat[(size_t)g * 256 + t];
  __syncthreads();
  for (int r = t; r < 1024; r += 256) {
    int n = r >> 8, o = r & 255;
    float s = 0.f;
    for (int c = 0; c < 64; ++c) s += lin_w[o * 64 + c] * fs[n * 64 + c];
    hs[r] = s;
  }
  __syncthreads();
  if (t < 16) {  // (node n, head hh)
    int n = t >> 2, hh = t & 3;
    float s = 0.f, d = 0.f;
    for (int c = 0; c < 64; ++c) {
      float hv = hs[n * 256 + hh * 64 + c];
      s += hv * att_src[hh * 64 + c];
      d += hv * att_dst[hh * 64 + c];
    }
    as[t] = s;
    ad[t] = d;
  }
  __syncthreads();
  if (t < 16) {  // softmax over source j for (dest i, head hh); adj all-true
    int i = t >> 2, hh = t & 3;
    float sc[4], mx = -1e30f;
    for (int j = 0; j < 4; ++j) {
      float x = ad[i * 4 + hh] + as[j * 4 + hh];
      x = x > 0.f ? x : 0.2f * x;  // leaky_relu(0.2)
      sc[j] = x;
      mx = fmaxf(mx, x);
    }
    float den = 0.f;
    for (int j = 0; j < 4; ++j) { sc[j] = __expf(sc[j] - mx); den += sc[j]; }
    for (int j = 0; j < 4; ++j) alpha[t][j] = sc[j] / den;
  }
  __syncthreads();
  for (int r = t; r < 1024; r += 256) {
    int i = r >> 8, o = r & 255, hh = o >> 6;
    float s = 0.f;
    for (int j = 0; j < 4; ++j) s += alpha[i * 4 + hh][j] * hs[j * 256 + o];
    gout[(size_t)g * 1024 + r] = s + bias[o];
  }
}

// ================== feature projection + positional enc ==============
__global__ __launch_bounds__(256) void featproj_pe_kernel(
    const float* __restrict__ gout, const float* __restrict__ w,
    const float* __restrict__ bias, float* __restrict__ x) {
  int tok = blockIdx.x;  // b*3+s
  int o = threadIdx.x;   // 0..255
  int s = tok % 3;
  const float* gi = gout + (size_t)tok * 1024;
  const float* wr = w + (size_t)o * 1024;
  float acc = bias[o];
  for (int k = 0; k < 1024; ++k) acc += gi[k] * wr[k];
  float freq = __expf(-(float)(o & ~1) * (logf(10000.0f) / 256.0f));
  float ang = (float)s * freq;
  acc += (o & 1) ? cosf(ang) : sinf(ang);
  x[(size_t)tok * 256 + o] = acc;
}

// ====================== generic small linear =========================
__global__ __launch_bounds__(256) void lin_kernel(const float* __restrict__ in,
                                                  const float* __restrict__ w,
                                                  const float* __restrict__ bias,
                                                  float* __restrict__ out,
                                                  int M, int N, int K, int relu) {
  int idx = blockIdx.x * blockDim.x + threadIdx.x;
  if (idx >= M * N) return;
  int m = idx / N, n = idx - m * N;
  const float* ir = in + (size_t)m * K;
  const float* wr = w + (size_t)n * K;
  float acc = bias ? bias[n] : 0.f;
  for (int k = 0; k < K; ++k) acc += ir[k] * wr[k];
  if (relu) acc = fmaxf(acc, 0.f);
  out[idx] = acc;
}

// ================== MHA over S=3 tokens, 4 heads =====================
__global__ __launch_bounds__(64) void attn_kernel(const float* __restrict__ qkv,
                                                  float* __restrict__ outa) {
  __shared__ float q[3][256], k[3][256], v[3][256];
  int b = blockIdx.x, t = threadIdx.x;
  for (int i = t; i < 3 * 256; i += 64) {
    int s = i >> 8, d = i & 255;
    const float* row = qkv + (size_t)(b * 3 + s) * 768;
    q[s][d] = row[d];
    k[s][d] = row[256 + d];
    v[s][d] = row[512 + d];
  }
  __syncthreads();
  if (t < 12) {  // (head h, query qi)
    int h = t / 3, qi = t - h * 3;
    float sc[3], mx = -1e30f;
    for (int kj = 0; kj < 3; ++kj) {
      float s = 0.f;
      for (int d = 0; d < 64; ++d) s += q[qi][h * 64 + d] * k[kj][h * 64 + d];
      s *= 0.125f;  // 1/sqrt(DH=64)
      sc[kj] = s;
      mx = fmaxf(mx, s);
    }
    float den = 0.f;
    for (int kj = 0; kj < 3; ++kj) { sc[kj] = __expf(sc[kj] - mx); den += sc[kj]; }
    for (int kj = 0; kj < 3; ++kj) sc[kj] /= den;
    for (int d = 0; d < 64; ++d) {
      float o = sc[0] * v[0][h * 64 + d] + sc[1] * v[1][h * 64 + d] + sc[2] * v[2][h * 64 + d];
      outa[(size_t)(b * 3 + qi) * 256 + h * 64 + d] = o;
    }
  }
}

// =============== residual add + post-LayerNorm (in place) ============
__global__ __launch_bounds__(256) void add_ln_kernel(float* __restrict__ x,
                                                     const float* __restrict__ y,
                                                     const float* __restrict__ g,
                                                     const float* __restrict__ b) {
  __shared__ float red[256];
  int tok = blockIdx.x, t = threadIdx.x;
  float v = x[(size_t)tok * 256 + t] + y[(size_t)tok * 256 + t];
  red[t] = v;
  __syncthreads();
  for (int off = 128; off > 0; off >>= 1) {
    if (t < off) red[t] += red[t + off];
    __syncthreads();
  }
  float mu = red[0] * (1.0f / 256.0f);
  __syncthreads();
  float d = v - mu;
  red[t] = d * d;
  __syncthreads();
  for (int off = 128; off > 0; off >>= 1) {
    if (t < off) red[t] += red[t + off];
    __syncthreads();
  }
  float var = red[0] * (1.0f / 256.0f);
  x[(size_t)tok * 256 + t] = d * rsqrtf(var + 1e-5f) * g[t] + b[t];
}

// ================ temporal attention pooling over S ==================
__global__ __launch_bounds__(256) void temporal_kernel(const float* __restrict__ x,
                                                       const float* __restrict__ ta_w,
                                                       const float* __restrict__ ta_b,
                                                       float* __restrict__ ctx) {
  __shared__ float sc[3];
  __shared__ float xr[3][256];
  int b = blockIdx.x, t = threadIdx.x;
  for (int s = 0; s < 3; ++s) xr[s][t] = x[(size_t)(b * 3 + s) * 256 + t];
  __syncthreads();
  if (t < 3) {
    float acc = ta_b[0];
    for (int d = 0; d < 256; ++d) acc += xr[t][d] * ta_w[d];
    sc[t] = acc;
  }
  __syncthreads();
  float mx = fmaxf(sc[0], fmaxf(sc[1], sc[2]));
  float e0 = __expf(sc[0] - mx), e1 = __expf(sc[1] - mx), e2 = __expf(sc[2] - mx);
  float inv = 1.0f / (e0 + e1 + e2);
  ctx[(size_t)b * 256 + t] = (e0 * xr[0][t] + e1 * xr[1][t] + e2 * xr[2][t]) * inv;
}

// ============================ launch =================================
extern "C" void kernel_launch(void* const* d_in, const int* in_sizes, int n_in,
                              void* d_out, int out_size, void* d_ws, size_t ws_size,
                              hipStream_t stream) {
  (void)in_sizes; (void)n_in; (void)out_size; (void)ws_size;
  const float* x_t = (const float*)d_in[0];
  const float* x_f = (const float*)d_in[1];
  const float* conv1_w = (const float*)d_in[2];
  const float* conv1_b = (const float*)d_in[3];
  const float *bn1_g = (const float*)d_in[4], *bn1_b = (const float*)d_in[5],
              *bn1_m = (const float*)d_in[6], *bn1_v = (const float*)d_in[7];
  const float* conv2_w = (const float*)d_in[8];
  const float* conv2_b = (const float*)d_in[9];
  const float *bn2_g = (const float*)d_in[10], *bn2_b = (const float*)d_in[11],
              *bn2_m = (const float*)d_in[12], *bn2_v = (const float*)d_in[13];
  const float* conv3_w = (const float*)d_in[14];
  const float* conv3_b = (const float*)d_in[15];
  const float *bn3_g = (const float*)d_in[16], *bn3_b = (const float*)d_in[17],
              *bn3_m = (const float*)d_in[18], *bn3_v = (const float*)d_in[19];
  const float* gat_lin_w = (const float*)d_in[20];
  const float* gat_att_src = (const float*)d_in[21];
  const float* gat_att_dst = (const float*)d_in[22];
  const float* gat_bias = (const float*)d_in[23];
  const float* feat_proj_w = (const float*)d_in[24];
  const float* feat_proj_b = (const float*)d_in[25];
  const float* tr_in_w = (const float*)d_in[26];
  const float* tr_in_b = (const float*)d_in[27];
  const float* tr_out_w = (const float*)d_in[28];
  const float* tr_out_b = (const float*)d_in[29];
  const float* tr_l1_w = (const float*)d_in[30];
  const float* tr_l1_b = (const float*)d_in[31];
  const float* tr_l2_w = (const float*)d_in[32];
  const float* tr_l2_b = (const float*)d_in[33];
  const float *tr_ln1_g = (const float*)d_in[34], *tr_ln1_b = (const float*)d_in[35];
  const float *tr_ln2_g = (const float*)d_in[36], *tr_ln2_b = (const float*)d_in[37];
  const float* ta_w = (const float*)d_in[38];
  const float* ta_b = (const float*)d_in[39];
  const float *fc1_w = (const float*)d_in[40], *fc1_b = (const float*)d_in[41];
  const float *fc2_w = (const float*)d_in[42], *fc2_b = (const float*)d_in[43];
  const float *fc3_w = (const float*)d_in[44], *fc3_b = (const float*)d_in[45];
  float* logits = (float*)d_out;

  // ---- workspace carve-up ----
  char* ws = (char*)d_ws;
  size_t cur = 0;
  auto take = [&](size_t bytes) -> void* {
    void* p = ws + cur;
    cur = (cur + bytes + 255) & ~(size_t)255;
    return p;
  };
  __bf16* xb  = (__bf16*)take((size_t)G_ * C_ * L_ * 2);    // branch input bf16
  __bf16* c1o = (__bf16*)take((size_t)G_ * 32 * 2048 * 2);  // conv1 out
  __bf16* c2o = (__bf16*)take((size_t)G_ * 64 * 1024 * 2);  // conv2 out
  __bf16* w1p = (__bf16*)take((size_t)2 * 32 * 32 * 2);
  __bf16* w2p = (__bf16*)take((size_t)2 * 64 * 224 * 2);
  __bf16* w3p = (__bf16*)take((size_t)2 * 128 * 192 * 2);
  float* s1 = (float*)take(2 * 32 * 4);
  float* h1 = (float*)take(2 * 32 * 4);
  float* s2 = (float*)take(2 * 64 * 4);
  float* h2 = (float*)take(2 * 64 * 4);
  float* s3 = (float*)take(2 * 128 * 4);
  float* h3 = (float*)take(2 * 128 * 4);
  float* featb = (float*)take((size_t)G_ * 256 * 4);   // pooled GAT features
  float* gof   = (float*)take((size_t)G_ * 1024 * 4);  // GAT output
  float* xbuf  = (float*)take((size_t)G_ * 256 * 4);   // transformer state
  float* qkvb  = (float*)take((size_t)G_ * 768 * 4);   // qkv / scratch
  float* tbuf  = (float*)take((size_t)G_ * 256 * 4);   // attention out
  float* ffb   = (float*)take((size_t)G_ * 1024 * 4);  // FFN hidden
  float* ctx   = (float*)take((size_t)B_ * 256 * 4);
  float* o1    = (float*)take((size_t)B_ * 128 * 4);
  float* o2    = (float*)take((size_t)B_ * 64 * 4);

  // ---- parameter prep (covers both branches) ----
  pack_w_swizzle_kernel<<<(2 * 32 * 32 + 255) / 256, 256, 0, stream>>>(conv1_w, w1p, 32, 2 * 15, 32);
  pack_w_swizzle_kernel<<<(2 * 64 * 224 + 255) / 256, 256, 0, stream>>>(conv2_w, w2p, 64, 32 * 7, 224);
  pack_w_swizzle_kernel<<<(2 * 128 * 192 + 255) / 256, 256, 0, stream>>>(conv3_w, w3p, 128, 64 * 3, 192);
  bn_prep_kernel<<<1, 64, 0, stream>>>(conv1_b, bn1_g, bn1_b, bn1_m, bn1_v, s1, h1, 64);
  bn_prep_kernel<<<1, 128, 0, stream>>>(conv2_b, bn2_g, bn2_b, bn2_m, bn2_v, s2, h2, 128);
  bn_prep_kernel<<<1, 256, 0, stream>>>(conv3_b, bn3_g, bn3_b, bn3_m, bn3_v, s3, h3, 256);
  zero_kernel<<<(G_ * 256 + 255) / 256, 256, 0, stream>>>(featb, G_ * 256);

  // ---- dual-branch CNN (time, freq) with WMMA implicit GEMM ----
  for (int br = 0; br < 2; ++br) {
    const float* xsrc = br ? x_f : x_t;
    cast_transpose_kernel<<<((size_t)G_ * C_ * L_ + 255) / 256, 256, 0, stream>>>(xsrc, xb);
    conv_wmma_kernel<2, 32, 15, 2, 7, 4096, 2048, 32, 4, false>
        <<<dim3(G_, 2048 / 64), 64, 0, stream>>>(xb, w1p + br * 32 * 32, s1 + br * 32,
                                                 h1 + br * 32, c1o, nullptr, 0, 0.f);
    conv_wmma_kernel<32, 64, 7, 2, 3, 2048, 1024, 224, 4, false>
        <<<dim3(G_, 1024 / 64), 128, 0, stream>>>(c1o, w2p + br * 64 * 224, s2 + br * 64,
                                                  h2 + br * 64, c2o, nullptr, 0, 0.f);
    conv_wmma_kernel<64, 128, 3, 1, 1, 1024, 1024, 192, 4, true>
        <<<dim3(G_, 1024 / 64), 256, 0, stream>>>(c2o, w3p + br * 128 * 192, s3 + br * 128,
                                                  h3 + br * 128, nullptr, featb + br * 128,
                                                  256, 1.0f / 1024.0f);
  }

  // ---- GAT over 4-node graphs + projection to transformer dim ----
  gat_kernel<<<G_, 256, 0, stream>>>(featb, gat_lin_w, gat_att_src, gat_att_dst, gat_bias, gof);
  featproj_pe_kernel<<<G_, 256, 0, stream>>>(gof, feat_proj_w, feat_proj_b, xbuf);

  // ---- 2-layer post-norm transformer encoder over S=3 tokens ----
  for (int l = 0; l < 2; ++l) {
    lin_kernel<<<(G_ * 768 + 255) / 256, 256, 0, stream>>>(
        xbuf, tr_in_w + (size_t)l * 768 * 256, tr_in_b + l * 768, qkvb, G_, 768, 256, 0);
    attn_kernel<<<B_, 64, 0, stream>>>(qkvb, tbuf);
    lin_kernel<<<(G_ * 256 + 255) / 256, 256, 0, stream>>>(
        tbuf, tr_out_w + (size_t)l * 256 * 256, tr_out_b + l * 256, qkvb, G_, 256, 256, 0);
    add_ln_kernel<<<G_, 256, 0, stream>>>(xbuf, qkvb, tr_ln1_g + l * 256, tr_ln1_b + l * 256);
    lin_kernel<<<(G_ * 1024 + 255) / 256, 256, 0, stream>>>(
        xbuf, tr_l1_w + (size_t)l * 1024 * 256, tr_l1_b + l * 1024, ffb, G_, 1024, 256, 1);
    lin_kernel<<<(G_ * 256 + 255) / 256, 256, 0, stream>>>(
        ffb, tr_l2_w + (size_t)l * 256 * 1024, tr_l2_b + l * 256, qkvb, G_, 256, 1024, 0);
    add_ln_kernel<<<G_, 256, 0, stream>>>(xbuf, qkvb, tr_ln2_g + l * 256, tr_ln2_b + l * 256);
  }

  // ---- temporal attention + classifier head ----
  temporal_kernel<<<B_, 256, 0, stream>>>(xbuf, ta_w, ta_b, ctx);
  lin_kernel<<<(B_ * 128 + 255) / 256, 256, 0, stream>>>(ctx, fc1_w, fc1_b, o1, B_, 128, 256, 1);
  lin_kernel<<<(B_ * 64 + 255) / 256, 256, 0, stream>>>(o1, fc2_w, fc2_b, o2, B_, 64, 128, 1);
  lin_kernel<<<(B_ * 5 + 255) / 256, 256, 0, stream>>>(o2, fc3_w, fc3_b, logits, B_, 5, 64, 0);
}
